// MeanGraphSage_79972291052240
// MI455X (gfx1250) — matrix operand order repaired and checked by reading, not compiled, not run on hardware
//
#include <hip/hip_runtime.h>

typedef __attribute__((ext_vector_type(16))) _Float16 v16h;
typedef __attribute__((ext_vector_type(8)))  _Float16 v8h;
typedef __attribute__((ext_vector_type(8)))  float    v8f;

#define D_FEAT 64
#define U2     64   // UNITS/2 (each kernel maps 64 -> 64)
#define BPITCH 72   // LDS pitch (halfs) for transposed weights: bank-friendly

// ---------------------------------------------------------------------------
// Kernel 1: zero the workspace (sums[N*64] ++ counts[N]) every call.
// ---------------------------------------------------------------------------
__global__ void zero_ws_kernel(float* __restrict__ ws, int n) {
  int i = blockIdx.x * blockDim.x + threadIdx.x;
  if (i < n) ws[i] = 0.0f;
}

// ---------------------------------------------------------------------------
// Kernel 2: edge scatter. One thread per (edge, 4-feature quad).
// sums[row] += x[col] * w ; counts[row] += 1. Atomics land in the 12.8MB
// L2-resident scratch region (x itself is 12.8MB -> L2 hits for gathers).
// ---------------------------------------------------------------------------
__global__ void edge_scatter_kernel(const float* __restrict__ x,
                                    const long long* __restrict__ ei,  // [2,E] int64
                                    const float* __restrict__ w,
                                    float* __restrict__ sums,
                                    float* __restrict__ counts,
                                    int E) {
  int idx = blockIdx.x * blockDim.x + threadIdx.x;
  if (idx >= (E << 4)) return;          // E*16 threads
  int e = idx >> 4;
  int q = idx & 15;                     // which float4 of the 64 features
  int dst = (int)ei[e];                 // row: target node
  int src = (int)ei[E + e];             // col: source neighbor
  float wt = w[e];
  float4 xv = *((const float4*)(x + (size_t)src * D_FEAT) + q);
  float* sp = sums + (size_t)dst * D_FEAT + (q << 2);
  atomicAdd(sp + 0, xv.x * wt);
  atomicAdd(sp + 1, xv.y * wt);
  atomicAdd(sp + 2, xv.z * wt);
  atomicAdd(sp + 3, xv.w * wt);
  if (q == 0) atomicAdd(counts + dst, 1.0f);
}

// ---------------------------------------------------------------------------
// Kernel 3: fused mean-normalize + dual 64x64 projection (WMMA f16->f32)
// + concat + bias + ReLU. 256 threads = 8 waves; each wave owns 16 nodes.
// 16 x v_wmma_f32_16x16x32_f16 per wave.
// ---------------------------------------------------------------------------
__global__ __launch_bounds__(256, 1)
void sage_wmma_kernel(const float* __restrict__ x,
                      const float* __restrict__ sums,
                      const float* __restrict__ counts,
                      const float* __restrict__ sk,     // self_kernel    [64,64]
                      const float* __restrict__ nk,     // neighbor_kernel[64,64]
                      const float* __restrict__ bias,   // [128]
                      float* __restrict__ out,          // [N,128]
                      int N) {
  // Stage both weight matrices TRANSPOSED (column-major, f16) in LDS so each
  // WMMA B operand is contiguous: kbT[mat][n*BPITCH + k].
  __shared__ _Float16 kbT[2][U2 * BPITCH];   // 2 * 9KB
  __shared__ float    bias_lds[128];
  for (int i = threadIdx.x; i < D_FEAT * U2; i += 256) {
    int k = i >> 6, n = i & 63;                 // global layout: [k][n]
    kbT[0][n * BPITCH + k] = (_Float16)sk[i];
    kbT[1][n * BPITCH + k] = (_Float16)nk[i];
  }
  for (int i = threadIdx.x; i < 128; i += 256) bias_lds[i] = bias[i];
  __syncthreads();

  const int lane   = threadIdx.x & 31;
  const int wave   = threadIdx.x >> 5;
  const int m_base = (blockIdx.x * 8 + wave) * 16;
  if (m_base >= N) return;              // wave-uniform exit: EXEC stays all-1s

  const int  mrow = lane & 15;          // M (A rows) / N (B,C cols) within tile
  const int  hi   = lane >> 4;          // lane group 0/1
  const int  row  = m_base + mrow;
  const int  row_c = row < N ? row : N - 1;   // clamp loads
  const bool full  = (m_base + 16 <= N);      // wave-uniform store fast-path
  const float inv  = __builtin_amdgcn_rcpf(fmaxf(counts[row_c], 1.0f));

  // --- Build A operands (16x32, contiguous K order, same as B):
  //     half h -> K = hi*16 + h. Each (kh, h/8) group is a run
  //     of 8 consecutive K -> two float4 loads per group.
  v16h a[2][2];   // [0]=self(x) / [1]=neighbor(mean), [K 0..31]/[K 32..63]
  const float* xr = x    + (size_t)row_c * D_FEAT;
  const float* nr = sums + (size_t)row_c * D_FEAT;
  #pragma unroll
  for (int g = 0; g < 4; ++g) {                 // g = (kh<<1) | half16
    const int kh  = g >> 1;
    const int hb  = (g & 1) << 3;               // half-index base within v16h
    const int Kb  = kh * 32 + (hi << 4) + ((g & 1) << 3);
    float4 x0 = *(const float4*)(xr + Kb);
    float4 x1 = *(const float4*)(xr + Kb + 4);
    float4 n0v = *(const float4*)(nr + Kb);
    float4 n1v = *(const float4*)(nr + Kb + 4);
    a[0][kh][hb + 0] = (_Float16)x0.x;  a[0][kh][hb + 1] = (_Float16)x0.y;
    a[0][kh][hb + 2] = (_Float16)x0.z;  a[0][kh][hb + 3] = (_Float16)x0.w;
    a[0][kh][hb + 4] = (_Float16)x1.x;  a[0][kh][hb + 5] = (_Float16)x1.y;
    a[0][kh][hb + 6] = (_Float16)x1.z;  a[0][kh][hb + 7] = (_Float16)x1.w;
    a[1][kh][hb + 0] = (_Float16)(n0v.x * inv); a[1][kh][hb + 1] = (_Float16)(n0v.y * inv);
    a[1][kh][hb + 2] = (_Float16)(n0v.z * inv); a[1][kh][hb + 3] = (_Float16)(n0v.w * inv);
    a[1][kh][hb + 4] = (_Float16)(n1v.x * inv); a[1][kh][hb + 5] = (_Float16)(n1v.y * inv);
    a[1][kh][hb + 6] = (_Float16)(n1v.z * inv); a[1][kh][hb + 7] = (_Float16)(n1v.w * inv);
  }

  // --- B (32x16): lane holds column N = mrow, halves h -> K = hi*16 + h.
  //     Transposed LDS => each v16h is 2 contiguous 16B DS loads.
  #pragma unroll
  for (int mat = 0; mat < 2; ++mat) {
    #pragma unroll
    for (int n0 = 0; n0 < U2; n0 += 16) {
      const _Float16* colp = &kbT[mat][(n0 + mrow) * BPITCH + (hi << 4)];
      v8h b0lo = *(const v8h*)(colp);
      v8h b0hi = *(const v8h*)(colp + 8);
      v8h b1lo = *(const v8h*)(colp + 32);
      v8h b1hi = *(const v8h*)(colp + 40);
      v16h b0 = __builtin_shufflevector(b0lo, b0hi, 0, 1, 2, 3, 4, 5, 6, 7,
                                        8, 9, 10, 11, 12, 13, 14, 15);
      v16h b1 = __builtin_shufflevector(b1lo, b1hi, 0, 1, 2, 3, 4, 5, 6, 7,
                                        8, 9, 10, 11, 12, 13, 14, 15);
      v8f c = {};
      c = __builtin_amdgcn_wmma_f32_16x16x32_f16(false, a[mat][0], false, b0,
                                                 (short)0, c, false, false);
      c = __builtin_amdgcn_wmma_f32_16x16x32_f16(false, a[mat][1], false, b1,
                                                 (short)0, c, false, false);
      // --- C layout: VGPR r -> M = r + 8*hi, N = mrow. Fuse bias + ReLU.
      int   col = mat * U2 + n0 + mrow;           // concat offset
      float bv  = bias_lds[col];
      float* op = out + (size_t)(m_base + (hi << 3)) * 128 + col;
      if (full) {                                  // wave-uniform branch
        #pragma unroll
        for (int r = 0; r < 8; ++r) {
          float v = c[r] + bv;
          op[r * 128] = v > 0.0f ? v : 0.0f;
        }
      } else {
        #pragma unroll
        for (int r = 0; r < 8; ++r) {
          if (m_base + r + (hi << 3) < N) {
            float v = c[r] + bv;
            op[r * 128] = v > 0.0f ? v : 0.0f;
          }
        }
      }
    }
  }
}

// ---------------------------------------------------------------------------
extern "C" void kernel_launch(void* const* d_in, const int* in_sizes, int n_in,
                              void* d_out, int out_size, void* d_ws, size_t ws_size,
                              hipStream_t stream) {
  const float*     x    = (const float*)d_in[0];
  const long long* ei   = (const long long*)d_in[1];   // int64 per reference
  const float*     w    = (const float*)d_in[2];
  const float*     sk   = (const float*)d_in[3];
  const float*     nk   = (const float*)d_in[4];
  const float*     bias = (const float*)d_in[5];
  float*           out  = (float*)d_out;

  const int N = in_sizes[0] / D_FEAT;   // 50000
  const int E = in_sizes[2];            // 800000

  float* sums   = (float*)d_ws;                  // [N, 64]
  float* counts = sums + (size_t)N * D_FEAT;     // [N]

  const int nz = N * D_FEAT + N;
  zero_ws_kernel<<<(nz + 255) / 256, 256, 0, stream>>>(sums, nz);

  const int total = E << 4;                      // E * 16 threads
  edge_scatter_kernel<<<(total + 255) / 256, 256, 0, stream>>>(
      x, ei, w, sums, counts, E);

  const int blocks = (N + 127) / 128;            // 8 waves * 16 nodes per block
  sage_wmma_kernel<<<blocks, 256, 0, stream>>>(
      x, sums, counts, sk, nk, bias, out, N);
}
